// KDELoss_48335561949712
// MI455X (gfx1250) — compile-verified, matrix operand order checked by reading, not run on hardware
//
#include <hip/hip_runtime.h>
#include <hip/hip_bf16.h>
#include <math.h>

typedef _Float16 v16h __attribute__((ext_vector_type(16)));
typedef _Float16 v8h  __attribute__((ext_vector_type(8)));
typedef float    v8f  __attribute__((ext_vector_type(8)));

#define NROWS   8192
#define DIM     768
#define KAPPA   5.0f
#define NSPLIT  8            // N-dimension split across blocks
#define NT_PER  (NROWS / 16 / NSPLIT)   // 64 N tiles per block
#define KSTEPS  (DIM / 32)   // 24 WMMA K-steps

// ---------------- Kernel 1: row L2-normalize, f32 -> f16 ----------------
__global__ void __launch_bounds__(256) kde_normalize(const float* __restrict__ x,
                                                     _Float16* __restrict__ nrm) {
    const int lane = threadIdx.x & 31;
    const int wave = threadIdx.x >> 5;
    const int row  = blockIdx.x * 8 + wave;

    const float* xr = x + (size_t)row * DIM;
    float v[24];
    float ss = 0.0f;
#pragma unroll
    for (int j = 0; j < 24; ++j) {
        v[j] = xr[lane + 32 * j];
        ss += v[j] * v[j];
    }
#pragma unroll
    for (int m = 16; m >= 1; m >>= 1) ss += __shfl_xor(ss, m, 32);
    const float inv = 1.0f / fmaxf(sqrtf(ss), 1e-12f);

    _Float16* nr = nrm + (size_t)row * DIM;
#pragma unroll
    for (int j = 0; j < 24; ++j) nr[lane + 32 * j] = (_Float16)(v[j] * inv);
}

// Issue one 24KB N-tile copy global->LDS via async-to-LDS (6 x b128 per thread).
__device__ __forceinline__ void async_copy_tile(const _Float16* __restrict__ src,
                                                const _Float16* dstLDS, int tid) {
    const uint64_t gbase = (uint64_t)(uintptr_t)src;
    const unsigned lbase = (unsigned)(uintptr_t)dstLDS;  // low 32 bits = LDS offset
#pragma unroll
    for (int i = 0; i < 6; ++i) {
        const unsigned chunk = (unsigned)(tid + 256 * i) * 16u;
        unsigned lds = lbase + chunk;
        uint64_t ga  = gbase + chunk;
        asm volatile("global_load_async_to_lds_b128 %0, %1, off"
                     :: "v"(lds), "v"(ga) : "memory");
    }
}

// ---- Kernel 2: fused Gram(nnT) * exp(kappa*) * row-sum via WMMA --------
// Block = 128 M rows (8 waves x one 16-row tile, A frags in VGPRs) and a
// 1/8 slice of the N tiles. B tiles double-buffered in LDS with
// GLOBAL_LOAD_ASYNC_TO_LDS_B128 overlapping the 24-deep WMMA chain.
__global__ void __launch_bounds__(256) kde_gram_exp_rowsum(const _Float16* __restrict__ nrm,
                                                           float* __restrict__ dpart) {
    __shared__ __align__(16) _Float16 bbuf[2][16 * DIM];  // 2 x 24 KB

    const int tid    = threadIdx.x;
    const int lane   = tid & 31;
    const int wave   = tid >> 5;
    const int mblk   = blockIdx.x >> 3;       // 0..63  (128-row M block)
    const int nsplit = blockIdx.x & 7;        // 0..7   (N slice)
    const int mt     = mblk * 8 + wave;       // this wave's 16-row M tile

    // f16 A-frag layout (16x32 MxK): lane L holds row M=L%16;
    // lanes 0-15: K 0..7 & 16..23, lanes 16-31: K 8..15 & 24..31.
    const int mrow  = lane & 15;
    const int kbase = (lane < 16) ? 0 : 8;

    // Load this wave's A strip (16 x 768) into 24 register fragments.
    v16h af[KSTEPS];
    {
        const _Float16* __restrict__ arow = nrm + ((size_t)mt * 16 + mrow) * DIM + kbase;
#pragma unroll
        for (int k = 0; k < KSTEPS; ++k) {
            v8h a0 = *(const v8h*)(arow + k * 32);
            v8h a1 = *(const v8h*)(arow + k * 32 + 16);
            af[k] = __builtin_shufflevector(a0, a1, 0, 1, 2, 3, 4, 5, 6, 7,
                                            8, 9, 10, 11, 12, 13, 14, 15);
        }
    }

    v8f rs = {0.f, 0.f, 0.f, 0.f, 0.f, 0.f, 0.f, 0.f};

    const int nt0 = nsplit * NT_PER;
    // Prime the pipeline: tile 0 -> buffer 0.
    async_copy_tile(nrm + (size_t)(nt0) * 16 * DIM, bbuf[0], tid);

    for (int nt = 0; nt < NT_PER; ++nt) {
        const int cur = nt & 1;
        if (nt + 1 < NT_PER) {
            // Overlap: fetch next tile into the other buffer...
            async_copy_tile(nrm + (size_t)(nt0 + nt + 1) * 16 * DIM, bbuf[cur ^ 1], tid);
            // ...then wait only for the PREVIOUS 6 ops (in-order completion).
            asm volatile("s_wait_asynccnt 6" ::: "memory");
        } else {
            asm volatile("s_wait_asynccnt 0" ::: "memory");
        }
        __syncthreads();   // all waves' slices of bbuf[cur] are in LDS

        const _Float16* __restrict__ brow = &bbuf[cur][mrow * DIM + kbase];
        v8f c = {0.f, 0.f, 0.f, 0.f, 0.f, 0.f, 0.f, 0.f};
#pragma unroll
        for (int k = 0; k < KSTEPS; ++k) {
            v8h b0 = *(const v8h*)(brow + k * 32);
            v8h b1 = *(const v8h*)(brow + k * 32 + 16);
            v16h b = __builtin_shufflevector(b0, b1, 0, 1, 2, 3, 4, 5, 6, 7,
                                             8, 9, 10, 11, 12, 13, 14, 15);
            // (neg_a, A, neg_b, B, c_mod, C, reuse_a, reuse_b)
            c = __builtin_amdgcn_wmma_f32_16x16x32_f16(false, af[k], false, b,
                                                       (short)0, c, false, false);
        }
#pragma unroll
        for (int r = 0; r < 8; ++r) rs[r] += __expf(KAPPA * c[r]);

        __syncthreads();   // tile consumed; its buffer may be overwritten next iter
    }

    // C layout: lanes 0-15 -> M=r, lanes 16-31 -> M=r+8; xor masks 8..1
    // reduce across the 16 N columns without crossing the halves.
    float* __restrict__ drow = dpart + (size_t)nsplit * NROWS + mt * 16;
#pragma unroll
    for (int r = 0; r < 8; ++r) {
        float s = rs[r];
#pragma unroll
        for (int m = 8; m >= 1; m >>= 1) s += __shfl_xor(s, m, 32);
        if (lane == 0)  drow[r] = s;
        if (lane == 16) drow[r + 8] = s;
    }
}

// -------- Kernel 3: loss = -mean(log(sum_p dpart[p] + eps)) -------------
__global__ void __launch_bounds__(256) kde_neglog_mean(const float* __restrict__ dpart,
                                                       float* __restrict__ out) {
    __shared__ float red[8];
    float s = 0.0f;
    for (int i = threadIdx.x; i < NROWS; i += 256) {
        float d = 0.0f;
#pragma unroll
        for (int p = 0; p < NSPLIT; ++p) d += dpart[(size_t)p * NROWS + i];
        s += __logf(d + 1e-9f);
    }
#pragma unroll
    for (int m = 16; m >= 1; m >>= 1) s += __shfl_xor(s, m, 32);
    if ((threadIdx.x & 31) == 0) red[threadIdx.x >> 5] = s;
    __syncthreads();
    if (threadIdx.x == 0) {
        float t = 0.0f;
#pragma unroll
        for (int i = 0; i < 8; ++i) t += red[i];
        out[0] = -(t / (float)NROWS);
    }
}

extern "C" void kernel_launch(void* const* d_in, const int* in_sizes, int n_in,
                              void* d_out, int out_size, void* d_ws, size_t ws_size,
                              hipStream_t stream) {
    const float* x = (const float*)d_in[0];
    float* out = (float*)d_out;

    _Float16* nrm = (_Float16*)d_ws;                                   // 12.58 MB
    float* dpart  = (float*)((char*)d_ws + (size_t)NROWS * DIM * 2);   // + 256 KB

    kde_normalize<<<NROWS / 8, 256, 0, stream>>>(x, nrm);
    kde_gram_exp_rowsum<<<64 * NSPLIT, 256, 0, stream>>>(nrm, dpart);
    kde_neglog_mean<<<1, 256, 0, stream>>>(dpart, out);
}
